// LSTMPointerNet_multitask_51539607552911
// MI455X (gfx1250) — compile-verified, matrix operand order checked by reading, not run on hardware
//
#include <hip/hip_runtime.h>
#include <hip/hip_bf16.h>

typedef __bf16 bf16_t;
typedef __attribute__((ext_vector_type(16))) __bf16 v16bf;
typedef __attribute__((ext_vector_type(8)))  __bf16 v8bf;
typedef __attribute__((ext_vector_type(8)))  float   v8f;

#define Bq    8
#define T1    129     // T+1
#define N1    257     // NS+1
#define Hd    512
#define G4    2048    // 4*H
#define MEMROWS 2064  // ceil16(8*257)
#define SIDEROWS 2048 // 8*256
#define QROWS 1040    // ceil16(8*129)
#define NB    4       // N-tiles per wave in the GEMM (A-fragment reuse)

// ---------------------------------------------------------------------------
// Weight packing into WMMA B-fragment layout (32x16 tiles, bf16).
// Layout: out[((nt*Ktiles + kt)*32 + lane)*16 + e]
//   n = nt*16 + (lane&15); k = kt*32 + (lane>>4)*8 + (e<8 ? e : e+8)
// ---------------------------------------------------------------------------
__global__ void pack_weight_bf16(const float* __restrict__ W, bf16_t* __restrict__ out,
                                 int K, int N, int transposeW)
{
    int idx = blockIdx.x * blockDim.x + threadIdx.x;
    if (idx >= K * N) return;
    int e    = idx & 15;
    int lane = (idx >> 4) & 31;
    int t    = idx >> 9;          // nt*Ktiles + kt
    int Ktiles = K >> 5;
    int kt = t % Ktiles;
    int nt = t / Ktiles;
    int n = nt * 16 + (lane & 15);
    int k = kt * 32 + (lane >> 4) * 8 + (e < 8 ? e : e + 8);
    float val = transposeW ? W[(size_t)n * K + k] : W[(size_t)k * N + n];
    out[idx] = (bf16_t)val;
}

// mem = concat(attn_mem, 0) + onehot(mem_sizes)*stop, rows padded to 2064, bf16
__global__ void build_mem_bf16(const float* __restrict__ attn_mem,
                               const int* __restrict__ mem_sizes,
                               const float* __restrict__ stop,
                               bf16_t* __restrict__ out)
{
    int idx = blockIdx.x * blockDim.x + threadIdx.x;
    if (idx >= MEMROWS * Hd) return;
    int d = idx & 511;
    int row = idx >> 9;
    float val = 0.f;
    if (row < Bq * N1) {
        int b = row / N1, n = row % N1;
        if (n < N1 - 1) val = attn_mem[((size_t)b * (N1 - 1) + n) * Hd + d];
        if (n == mem_sizes[b]) val += stop[d];
    }
    out[idx] = (bf16_t)val;
}

__global__ void conv_bf16(const float* __restrict__ in, bf16_t* __restrict__ out, int total)
{
    int idx = blockIdx.x * blockDim.x + threadIdx.x;
    if (idx >= total) return;
    out[idx] = (bf16_t)in[idx];
}

// x = concat(init_i, lstm_in) per batch, rows padded to 1040, bf16
__global__ void build_x_bf16(const float* __restrict__ lstm_in,
                             const float* __restrict__ init_i,
                             bf16_t* __restrict__ out)
{
    int idx = blockIdx.x * blockDim.x + threadIdx.x;
    if (idx >= QROWS * Hd) return;
    int d = idx & 511;
    int row = idx >> 9;
    float v = 0.f;
    if (row < Bq * T1) {
        int b = row / T1, t = row % T1;
        v = (t == 0) ? init_i[d] : lstm_in[((size_t)b * (T1 - 1) + (t - 1)) * Hd + d];
    }
    out[idx] = (bf16_t)v;
}

// h/c state init + zero bf16 pad rows of the query-like buffers
__global__ void init_state(const float* __restrict__ init_h, const float* __restrict__ init_c,
                           bf16_t* __restrict__ hA, bf16_t* __restrict__ hB,
                           float* __restrict__ cstate,
                           bf16_t* __restrict__ qbf, bf16_t* __restrict__ q2bf,
                           bf16_t* __restrict__ sebf)
{
    int idx = blockIdx.x * blockDim.x + threadIdx.x;
    if (idx >= 16 * Hd) return;
    int d = idx & 511;
    int r = idx >> 9;
    float h = (r < 8) ? init_h[d] : 0.f;
    hA[idx] = (bf16_t)h;
    hB[idx] = (bf16_t)h;
    if (r < 8) {
        cstate[r * Hd + d] = init_c[d];
        size_t p = (size_t)(Bq * T1 + r) * Hd + d;   // rows 1032..1039
        qbf[p]  = (bf16_t)0.f;
        q2bf[p] = (bf16_t)0.f;
        sebf[p] = (bf16_t)0.f;
    }
}

// ---------------------------------------------------------------------------
// bf16 WMMA GEMM: C[M,N] = A[Mpad,K] (row-major bf16) x Bpacked + bias.
// Each wave owns one 16-row M-tile and NB=4 consecutive 16-col N-tiles:
// one A fragment feeds 4 v_wmma_f32_16x16x32_bf16 accumulators per k-step,
// cutting vmem issue per wmma from 4x b128 to 2.5x b128.
// ---------------------------------------------------------------------------
__global__ void gemm_bf16_wmma(const bf16_t* __restrict__ A, const bf16_t* __restrict__ Bp,
                               float* __restrict__ C,
                               int Mtiles, int Ntiles, int Ktiles, int K, int N,
                               const float* __restrict__ bias1, const float* __restrict__ bias2)
{
    int wave = (blockIdx.x * blockDim.x + threadIdx.x) >> 5;
    int lane = threadIdx.x & 31;
    int Ngroups = Ntiles >> 2;                 // Ntiles is a multiple of NB=4
    if (wave >= Mtiles * Ngroups) return;      // wave-uniform: EXEC stays all-1s
    int tm  = wave / Ngroups;
    int tn0 = (wave % Ngroups) * NB;
    int khalf = lane >> 4;
    int mrow = tm * 16 + (lane & 15);

    v8f acc[NB];
#pragma unroll
    for (int nb = 0; nb < NB; ++nb) acc[nb] = (v8f){};

    const bf16_t* arow  = A + (size_t)mrow * K + khalf * 8;
    const bf16_t* bbase = Bp + ((size_t)tn0 * Ktiles * 32 + lane) * 16;
    const size_t bTile  = (size_t)Ktiles * 32 * 16;   // elems per packed N-tile

    for (int kt = 0; kt < Ktiles; ++kt) {
        v8bf alo = *(const v8bf*)(arow + kt * 32);
        v8bf ahi = *(const v8bf*)(arow + kt * 32 + 16);
        v16bf a;
#pragma unroll
        for (int i = 0; i < 8; ++i) { a[i] = alo[i]; a[i + 8] = ahi[i]; }
#pragma unroll
        for (int nb = 0; nb < NB; ++nb) {
            v16bf b = *(const v16bf*)(bbase + (size_t)nb * bTile + (size_t)kt * 32 * 16);
            acc[nb] = __builtin_amdgcn_wmma_f32_16x16x32_bf16(false, a, false, b,
                                                              (short)0, acc[nb], false, false);
        }
    }

#pragma unroll
    for (int nb = 0; nb < NB; ++nb) {
        int col = (tn0 + nb) * 16 + (lane & 15);
        float bv = 0.f;
        if (bias1) bv += bias1[col];
        if (bias2) bv += bias2[col];
#pragma unroll
        for (int v = 0; v < 8; ++v) {
            int row = tm * 16 + v + khalf * 8;
            C[(size_t)row * N + col] = acc[nb][v] + bv;
        }
    }
}

// ---------------------------------------------------------------------------
// One LSTM time step. 32 waves; wave jt owns hidden cols [jt*16, jt*16+16) and
// computes all 4 gates with 4 WMMA accumulators sharing one A fragment
// (h-state, 16x512 bf16, ping-pong hin->hout). Accumulators are seeded with
// xw (x@w_ih^T + b_ih + b_hh), so D = xw + h@w_hh^T directly.
// ---------------------------------------------------------------------------
__global__ void lstm_step(const float* __restrict__ xw, const bf16_t* __restrict__ whh_p,
                          const bf16_t* __restrict__ hin, bf16_t* __restrict__ hout,
                          float* __restrict__ cstate,
                          float* __restrict__ query, bf16_t* __restrict__ query_bf, int t)
{
    int wave = (blockIdx.x * blockDim.x + threadIdx.x) >> 5;   // 0..31
    int lane = threadIdx.x & 31;
    int jt = wave;
    int khalf = lane >> 4;
    int col = lane & 15;

    v8f acc[4];
#pragma unroll
    for (int g = 0; g < 4; ++g) {
#pragma unroll
        for (int v = 0; v < 8; ++v) {
            int r = v + khalf * 8;
            int rc = (r < 8) ? r : 7;
            int n = g * Hd + jt * 16 + col;
            float xv = xw[((size_t)rc * T1 + t) * G4 + n];
            acc[g][v] = (r < 8) ? xv : 0.f;
        }
    }

    const bf16_t* arow = hin + (size_t)(lane & 15) * Hd + khalf * 8;
    const int Ktiles = Hd / 32;
    for (int kt = 0; kt < Ktiles; ++kt) {
        v8bf alo = *(const v8bf*)(arow + kt * 32);
        v8bf ahi = *(const v8bf*)(arow + kt * 32 + 16);
        v16bf a;
#pragma unroll
        for (int i = 0; i < 8; ++i) { a[i] = alo[i]; a[i + 8] = ahi[i]; }
#pragma unroll
        for (int g = 0; g < 4; ++g) {
            int nt = g * 32 + jt;  // N-tile in the 2048-wide packed w_hh^T
            v16bf b = *(const v16bf*)(whh_p + ((size_t)(nt * Ktiles + kt) * 32 + lane) * 16);
            acc[g] = __builtin_amdgcn_wmma_f32_16x16x32_bf16(false, a, false, b,
                                                             (short)0, acc[g], false, false);
        }
    }

#pragma unroll
    for (int v = 0; v < 8; ++v) {
        int r = v + khalf * 8;
        if (r < 8) {
            int j = jt * 16 + col;
            float gi = 1.f / (1.f + __expf(-acc[0][v]));
            float gf = 1.f / (1.f + __expf(-acc[1][v]));
            float gg = tanhf(acc[2][v]);
            float go = 1.f / (1.f + __expf(-acc[3][v]));
            float cold = cstate[r * Hd + j];
            float cn = gf * cold + gi * gg;
            float h = go * tanhf(cn);
            cstate[r * Hd + j] = cn;
            size_t qrow = (size_t)r * T1 + t;
            query[qrow * Hd + j] = h;
            query_bf[qrow * Hd + j] = (bf16_t)h;
            hout[r * Hd + j] = (bf16_t)h;
        }
    }
}

// ---------------------------------------------------------------------------
// Fused Bahdanau attention: scores = tanh(feat + qadd (+qadd2)) . v, mask,
// softmax, context = p @ feat.  One block per (b,q). If score_out != null,
// writes raw scores (final unmasked pointer output) and skips softmax/ctx.
// feat stays resident in the 192MB L2 across the 1032 blocks.
// ---------------------------------------------------------------------------
__global__ void attn_fused(const float* __restrict__ feat, int N,
                           const float* __restrict__ qadd, const float* __restrict__ qadd2,
                           const float* __restrict__ vvec,
                           const int* __restrict__ sizes, int size_add,
                           float* __restrict__ ctx, bf16_t* __restrict__ ctx_bf,
                           float* __restrict__ score_out)
{
    __shared__ float s_sc[260];
    __shared__ float s_red[8];
    int b = blockIdx.x / T1;
    int q = blockIdx.x % T1;
    int tid = threadIdx.x;
    int wave = tid >> 5, lane = tid & 31;
    size_t qrow = (size_t)b * T1 + q;
    int msz = sizes ? (sizes[b] + size_add) : N;
    const float* qa  = qadd + qrow * Hd;
    const float* qa2 = qadd2 ? qadd2 + qrow * Hd : nullptr;

    // phase 1: scores, wave-per-n
    for (int n = wave; n < N; n += 8) {
        const float* frow = feat + ((size_t)b * N + n) * Hd;
        float part = 0.f;
#pragma unroll 4
        for (int i = 0; i < 16; ++i) {
            int h = lane * 16 + i;
            float s = frow[h] + qa[h];
            if (qa2) s += qa2[h];
            part += tanhf(s) * vvec[h];
        }
#pragma unroll
        for (int off = 16; off; off >>= 1) part += __shfl_xor(part, off, 32);
        if (lane == 0) s_sc[n] = (sizes && n >= msz) ? -1e18f : part;
    }
    __syncthreads();

    if (score_out) {
        for (int n = tid; n < N; n += 256)
            score_out[((size_t)b * T1 + q) * N + n] = s_sc[n];
        return;
    }

    // phase 2: softmax (max)
    float lmax = -1e30f;
    for (int n = tid; n < N; n += 256) lmax = fmaxf(lmax, s_sc[n]);
#pragma unroll
    for (int off = 16; off; off >>= 1) lmax = fmaxf(lmax, __shfl_xor(lmax, off, 32));
    if (lane == 0) s_red[wave] = lmax;
    __syncthreads();
    float gmax = s_red[0];
#pragma unroll
    for (int w = 1; w < 8; ++w) gmax = fmaxf(gmax, s_red[w]);
    __syncthreads();

    // phase 2b: exp + sum
    float lsum = 0.f;
    for (int n = tid; n < N; n += 256) {
        float e = __expf(s_sc[n] - gmax);
        s_sc[n] = e;
        lsum += e;
    }
#pragma unroll
    for (int off = 16; off; off >>= 1) lsum += __shfl_xor(lsum, off, 32);
    if (lane == 0) s_red[wave] = lsum;
    __syncthreads();
    float gsum = 0.f;
#pragma unroll
    for (int w = 0; w < 8; ++w) gsum += s_red[w];
    float inv = 1.f / gsum;
    __syncthreads();

    // phase 3: context
    for (int h = tid; h < Hd; h += 256) {
        float acc = 0.f;
        for (int n = 0; n < msz; ++n)
            acc += s_sc[n] * feat[((size_t)b * N + n) * Hd + h];
        float val = acc * inv;
        ctx[qrow * Hd + h] = val;
        ctx_bf[qrow * Hd + h] = (bf16_t)val;
    }
}

// ---------------------------------------------------------------------------
extern "C" void kernel_launch(void* const* d_in, const int* in_sizes, int n_in,
                              void* d_out, int out_size, void* d_ws, size_t ws_size,
                              hipStream_t stream)
{
    const float* attn_mem  = (const float*)d_in[0];
    const int*   mem_sizes = (const int*)  d_in[1];
    const float* lstm_in   = (const float*)d_in[2];
    const float* side_mem  = (const float*)d_in[3];
    const int*   side_sz   = (const int*)  d_in[4];
    const float* init_h    = (const float*)d_in[5];
    const float* init_c    = (const float*)d_in[6];
    const float* init_i    = (const float*)d_in[7];
    const float* stop      = (const float*)d_in[8];
    const float* attn_wm   = (const float*)d_in[9];
    const float* attn_wq   = (const float*)d_in[10];
    const float* attn_v    = (const float*)d_in[11];
    const float* hop_wm    = (const float*)d_in[12];
    const float* hop_wq    = (const float*)d_in[13];
    const float* hop_v     = (const float*)d_in[14];
    const float* side_wm   = (const float*)d_in[15];
    const float* side_wq   = (const float*)d_in[16];
    const float* side_v    = (const float*)d_in[17];
    const float* attn_ws   = (const float*)d_in[18];
    const float* w_ih      = (const float*)d_in[19];
    const float* w_hh      = (const float*)d_in[20];
    const float* b_ih      = (const float*)d_in[21];
    const float* b_hh      = (const float*)d_in[22];

    char* base = (char*)d_ws;
    size_t off = 0;
    auto alloc = [&](size_t bytes) -> char* {
        char* p = base + off;
        off = (off + bytes + 255) & ~size_t(255);
        return p;
    };

    float* attn_feat = (float*)alloc((size_t)MEMROWS * Hd * 4);
    float* hop_feat  = (float*)alloc((size_t)MEMROWS * Hd * 4);
    float* side_feat = (float*)alloc((size_t)SIDEROWS * Hd * 4);
    float* xw        = (float*)alloc((size_t)QROWS * G4 * 4);
    float* query     = (float*)alloc((size_t)QROWS * Hd * 4);
    float* q2        = (float*)alloc((size_t)QROWS * Hd * 4);
    float* sideE     = (float*)alloc((size_t)QROWS * Hd * 4);
    float* qwq       = (float*)alloc((size_t)QROWS * Hd * 4);
    float* cstate    = (float*)alloc((size_t)Bq * Hd * 4);
    bf16_t* mem_bf   = (bf16_t*)alloc((size_t)MEMROWS * Hd * 2);
    bf16_t* side_bf  = (bf16_t*)alloc((size_t)SIDEROWS * Hd * 2);
    bf16_t* x_bf     = (bf16_t*)alloc((size_t)QROWS * Hd * 2);
    bf16_t* query_bf = (bf16_t*)alloc((size_t)QROWS * Hd * 2);
    bf16_t* q2_bf    = (bf16_t*)alloc((size_t)QROWS * Hd * 2);
    bf16_t* sideE_bf = (bf16_t*)alloc((size_t)QROWS * Hd * 2);
    bf16_t* hA       = (bf16_t*)alloc((size_t)16 * Hd * 2);
    bf16_t* hB       = (bf16_t*)alloc((size_t)16 * Hd * 2);
    bf16_t* awm_p    = (bf16_t*)alloc((size_t)Hd * Hd * 2);
    bf16_t* hwm_p    = (bf16_t*)alloc((size_t)Hd * Hd * 2);
    bf16_t* swm_p    = (bf16_t*)alloc((size_t)Hd * Hd * 2);
    bf16_t* hwq_p    = (bf16_t*)alloc((size_t)Hd * Hd * 2);
    bf16_t* swq_p    = (bf16_t*)alloc((size_t)Hd * Hd * 2);
    bf16_t* awq_p    = (bf16_t*)alloc((size_t)Hd * Hd * 2);
    bf16_t* aws_p    = (bf16_t*)alloc((size_t)Hd * Hd * 2);
    bf16_t* wih_p    = (bf16_t*)alloc((size_t)Hd * G4 * 2);
    bf16_t* whh_p    = (bf16_t*)alloc((size_t)Hd * G4 * 2);
    // Late-stage projections reuse the dead xw region (8.5MB >= 3 * 2.13MB)
    float* qwq2 = xw;
    float* qA   = xw + (size_t)QROWS * Hd;
    float* sA   = xw + (size_t)2 * QROWS * Hd;

    const int TB = 256;
    auto grid1 = [](long total, int tb) { return (int)((total + tb - 1) / tb); };

    // --- pack weights (bf16 fragment layout) ---
    pack_weight_bf16<<<grid1((long)Hd * Hd, TB), TB, 0, stream>>>(attn_wm, awm_p, Hd, Hd, 0);
    pack_weight_bf16<<<grid1((long)Hd * Hd, TB), TB, 0, stream>>>(hop_wm,  hwm_p, Hd, Hd, 0);
    pack_weight_bf16<<<grid1((long)Hd * Hd, TB), TB, 0, stream>>>(side_wm, swm_p, Hd, Hd, 0);
    pack_weight_bf16<<<grid1((long)Hd * Hd, TB), TB, 0, stream>>>(hop_wq,  hwq_p, Hd, Hd, 0);
    pack_weight_bf16<<<grid1((long)Hd * Hd, TB), TB, 0, stream>>>(side_wq, swq_p, Hd, Hd, 0);
    pack_weight_bf16<<<grid1((long)Hd * Hd, TB), TB, 0, stream>>>(attn_wq, awq_p, Hd, Hd, 0);
    pack_weight_bf16<<<grid1((long)Hd * Hd, TB), TB, 0, stream>>>(attn_ws, aws_p, Hd, Hd, 0);
    pack_weight_bf16<<<grid1((long)Hd * G4, TB), TB, 0, stream>>>(w_ih, wih_p, Hd, G4, 1);
    pack_weight_bf16<<<grid1((long)Hd * G4, TB), TB, 0, stream>>>(w_hh, whh_p, Hd, G4, 1);

    // --- build activations ---
    build_mem_bf16<<<grid1((long)MEMROWS * Hd, TB), TB, 0, stream>>>(attn_mem, mem_sizes, stop, mem_bf);
    conv_bf16<<<grid1((long)SIDEROWS * Hd, TB), TB, 0, stream>>>(side_mem, side_bf, SIDEROWS * Hd);
    build_x_bf16<<<grid1((long)QROWS * Hd, TB), TB, 0, stream>>>(lstm_in, init_i, x_bf);
    init_state<<<grid1(16 * Hd, TB), TB, 0, stream>>>(init_h, init_c, hA, hB, cstate,
                                                      query_bf, q2_bf, sideE_bf);

    auto launch_gemm = [&](const bf16_t* A, const bf16_t* Bp, float* C,
                           int Mtiles, int N, const float* bi1, const float* bi2) {
        int Ntiles = N / 16, Ktiles = Hd / 32;
        long waves = (long)Mtiles * (Ntiles / NB);
        gemm_bf16_wmma<<<grid1(waves * 32, TB), TB, 0, stream>>>(
            A, Bp, C, Mtiles, Ntiles, Ktiles, Hd, N, bi1, bi2);
    };

    // --- projections ---
    launch_gemm(mem_bf,  awm_p, attn_feat, MEMROWS / 16, Hd, nullptr, nullptr);
    launch_gemm(mem_bf,  hwm_p, hop_feat,  MEMROWS / 16, Hd, nullptr, nullptr);
    launch_gemm(side_bf, swm_p, side_feat, SIDEROWS / 16, Hd, nullptr, nullptr);
    launch_gemm(x_bf,    wih_p, xw,        QROWS / 16,   G4, b_ih, b_hh);

    // --- sequential LSTM scan (129 steps, ping-pong h state) ---
    for (int t = 0; t < T1; ++t) {
        const bf16_t* hin = (t & 1) ? hB : hA;
        bf16_t* hout      = (t & 1) ? hA : hB;
        lstm_step<<<4, 256, 0, stream>>>(xw, whh_p, hin, hout, cstate, query, query_bf, t);
    }

    // --- hop attention ---
    launch_gemm(query_bf, hwq_p, qwq, QROWS / 16, Hd, nullptr, nullptr);
    attn_fused<<<Bq * T1, 256, 0, stream>>>(hop_feat, N1, qwq, nullptr, hop_v,
                                            mem_sizes, 1, q2, q2_bf, nullptr);

    // --- side attention ---
    launch_gemm(q2_bf, swq_p, qwq2, QROWS / 16, Hd, nullptr, nullptr);
    attn_fused<<<Bq * T1, 256, 0, stream>>>(side_feat, N1 - 1, qwq2, nullptr, side_v,
                                            side_sz, 0, sideE, sideE_bf, nullptr);

    // --- final pointer scores ---
    launch_gemm(q2_bf,    awq_p, qA, QROWS / 16, Hd, nullptr, nullptr);
    launch_gemm(sideE_bf, aws_p, sA, QROWS / 16, Hd, nullptr, nullptr);
    attn_fused<<<Bq * T1, 256, 0, stream>>>(attn_feat, N1, qA, sA, attn_v,
                                            nullptr, 0, nullptr, nullptr, (float*)d_out);
}
